// STA_LSTM_67697274519666
// MI455X (gfx1250) — matrix operand order repaired and checked by reading, not compiled, not run on hardware
//
#include <hip/hip_runtime.h>
#include <hip/hip_bf16.h>

// ---------------------------------------------------------------------------
// STA-LSTM for MI455X (gfx1250): bf16 WMMA GEMMs, TDM tensor_load_to_lds tile
// staging (TENSORcnt), async-to-LDS staging (ASYNCcnt), fused LSTM step.
// Sizes (fixed by reference): B=256, T=64, D=128, H=512, O=128, IN=8192, 4H=2048
// ---------------------------------------------------------------------------

typedef __bf16 bf16_t;
typedef __attribute__((ext_vector_type(16))) __bf16 v16bf;
typedef __attribute__((ext_vector_type(8)))  __bf16 v8bf;
typedef __attribute__((ext_vector_type(8)))  float  v8f;
typedef __attribute__((ext_vector_type(4)))  unsigned int v4u;
typedef __attribute__((ext_vector_type(8)))  int v8i_t;
typedef __attribute__((ext_vector_type(4)))  int v4i_t;

#define B_SZ  256
#define T_SZ  64
#define D_SZ  128
#define H_SZ  512
#define O_SZ  128
#define IN_SZ 8192
#define G4_SZ 2048

// ---- CDNA5 async memory->LDS copy (tracked by ASYNCcnt) --------------------
// Generic pointers to LDS carry the LDS byte offset in addr[31:0] (ISA 10.2).
// IOFFSET applies to BOTH the LDS and global address (ISA 8 §4.4), so chunked
// copies reuse one base address with an immediate offset.
template <int OFF>
__device__ __forceinline__ void async_copy_b128(const bf16_t* g, bf16_t* l) {
    unsigned lds = (unsigned)(unsigned long long)l;
    asm volatile("global_load_async_to_lds_b128 %0, %1, off offset:%2"
                 :: "v"(lds), "v"(g), "i"(OFF) : "memory");
}
__device__ __forceinline__ void wait_async_all() {
    asm volatile("s_wait_asynccnt 0x0" ::: "memory");
}

// ---- CDNA5 Tensor Data Mover: 2-D tile DMA global->LDS (TENSORcnt) ---------
// D# packing per ISA 8 §8.3/8.4: group0 = {count/flags, lds_addr, global_addr,
// type=2}; group1 = {data_size, tensor_dim0/1, tile_dim0/1/2, dim strides}.
// data_size=1 (2-byte elements). Tile stored packed row-major in LDS
// (row stride = tile_cols elements), matching the As/Bs layouts below.
// This toolchain exposes the 6-arg builtin (extra int32x8, zero-filled here).
__device__ __forceinline__ void tdm_load_2d(const bf16_t* g, bf16_t* l,
                                            unsigned tile_cols, unsigned tile_rows,
                                            unsigned tensor_cols /* row stride, elems */)
{
    unsigned long long ga = (unsigned long long)g;
    unsigned lds = (unsigned)(unsigned long long)l;
    v4u g0;
    g0[0] = 1u;                                               // count=1, user mode
    g0[1] = lds;                                              // lds_addr (bytes)
    g0[2] = (unsigned)ga;                                     // global_addr[31:0]
    g0[3] = (unsigned)((ga >> 32) & 0x01FFFFFFu) | (2u << 30); // addr[56:32] | type=2
    unsigned long long s0 = tensor_cols;                      // tensor_dim0_stride
    v8i_t g1;
    g1[0] = (int)(1u << 16);                                  // data_size=1 (2B)
    g1[1] = (int)((tensor_cols & 0xFFFFu) << 16);             // tensor_dim0[15:0]
    g1[2] = (int)((tensor_cols >> 16) | ((tile_rows & 0xFFFFu) << 16)); // td0[31:16]|td1[15:0]
    g1[3] = (int)(((tile_rows >> 16) & 0xFFFFu) | (tile_cols << 16));   // td1[31:16]|tile_dim0
    g1[4] = (int)(tile_rows & 0xFFFFu);                       // tile_dim1 (tile_dim2=0)
    g1[5] = (int)(s0 & 0xFFFFFFFFu);                          // stride0[31:0]
    g1[6] = (int)((s0 >> 32) & 0xFFFFu);                      // stride0[47:32] (stride1=0)
    g1[7] = 0;
    v4i_t gz4 = {0, 0, 0, 0};                                 // groups 2/3 unused (2-D)
    v8i_t gz8 = {0, 0, 0, 0, 0, 0, 0, 0};
    __builtin_amdgcn_tensor_load_to_lds(g0, g1, gz4, gz4, gz8, 0);
}
__device__ __forceinline__ void wait_tensor_all() {
    __builtin_amdgcn_s_wait_tensorcnt(0);
}

// ---- WMMA fragment helpers (wave32; this code's contiguous k order, A and B alike) -----------
// A 16x32 bf16: lanes 0-15 hold row M=lane, K = 0..15;
//               lanes 16-31 hold row M=lane-16, K = 16..31.
__device__ __forceinline__ v16bf frag_a(const bf16_t* As, int ld, int mbase, int kk, int lane) {
    int lidx = lane & 15, hsel = lane >> 4;
    const bf16_t* r = As + (mbase + lidx) * ld + kk + hsel * 16;
    v8bf lo = *(const v8bf*)(r);
    v8bf hi = *(const v8bf*)(r + 8);
    return __builtin_shufflevector(lo, hi, 0,1,2,3,4,5,6,7,8,9,10,11,12,13,14,15);
}
// B 32x16 bf16 (staged transposed in LDS as Bt[n][k]): lanes 0-15 hold
// column N=lane, K=0..15; lanes 16-31 hold column N=lane-16, K=16..31.
__device__ __forceinline__ v16bf frag_b(const bf16_t* Bt, int ld, int nbase, int kk, int lane) {
    int lidx = lane & 15, hsel = lane >> 4;
    return *(const v16bf*)(Bt + (nbase + lidx) * ld + kk + hsel * 16);
}
__device__ __forceinline__ v8f wmma_bf16(v16bf a, v16bf b, v8f c) {
    return __builtin_amdgcn_wmma_f32_16x16x32_bf16(false, a, false, b, (short)0, c, false, false);
}
__device__ __forceinline__ float sigmoidf_(float x) { return 1.0f / (1.0f + __expf(-x)); }

// ---------------------------------------------------------------------------
// Generic bf16 GEMM: C[M,N] = A[M,K] @ W[N,K]^T (+ bias0 + bias1), f32 out.
// Block: 256 thr = 8 waves as 2(M)x4(N); block tile 64x128, K-step 64,
// double-buffered LDS; tiles staged by the Tensor Data Mover (wave 0 issues
// one 2-D descriptor per tile), overlapped with WMMA compute.
// Requires K % 64 == 0 (true here: 8192, 128).
// ---------------------------------------------------------------------------
__global__ __launch_bounds__(256) void gemm_bf16_wmma(
    const unsigned short* __restrict__ A_, const unsigned short* __restrict__ W_,
    float* __restrict__ Cout, const float* __restrict__ bias0,
    const float* __restrict__ bias1, int M, int N, int K)
{
    const bf16_t* A = (const bf16_t*)A_;
    const bf16_t* W = (const bf16_t*)W_;
    __shared__ bf16_t As[2][64 * 64];    // 2 x 8KB,  A tile  [m][k]
    __shared__ bf16_t Bs[2][128 * 64];   // 2 x 16KB, W tile transposed [n][k]
    const int tid  = threadIdx.x;
    const int wave = tid >> 5, lane = tid & 31;
    const int m0 = blockIdx.x * 64;
    const int n0 = blockIdx.y * 128;
    const int wm = (wave >> 2) * 32;  // 0 / 32
    const int wn = (wave & 3) * 32;   // 0..96

    const bf16_t* abase = A + (size_t)m0 * K;   // tile row origin
    const bf16_t* wbase = W + (size_t)n0 * K;

    auto stage_tdm = [&](int buf, int k0) {     // whole-tile DMA, one wave
        tdm_load_2d(abase + k0, &As[buf][0], 64, 64, (unsigned)K);
        tdm_load_2d(wbase + k0, &Bs[buf][0], 64, 128, (unsigned)K);
    };

    v8f acc[2][2] = {};

    if (wave == 0) { stage_tdm(0, 0); wait_tensor_all(); }
    __syncthreads();

    const int niter = K >> 6;
    for (int it = 0; it < niter; ++it) {
        const int buf = it & 1;
        if (wave == 0 && it + 1 < niter) stage_tdm(buf ^ 1, (it + 1) << 6); // overlap DMA
#pragma unroll
        for (int kk = 0; kk < 64; kk += 32) {
            v16bf a0 = frag_a(&As[buf][0], 64, wm,      kk, lane);
            v16bf a1 = frag_a(&As[buf][0], 64, wm + 16, kk, lane);
            v16bf b0 = frag_b(&Bs[buf][0], 64, wn,      kk, lane);
            v16bf b1 = frag_b(&Bs[buf][0], 64, wn + 16, kk, lane);
            acc[0][0] = wmma_bf16(a0, b0, acc[0][0]);
            acc[0][1] = wmma_bf16(a0, b1, acc[0][1]);
            acc[1][0] = wmma_bf16(a1, b0, acc[1][0]);
            acc[1][1] = wmma_bf16(a1, b1, acc[1][1]);
        }
        if (wave == 0) wait_tensor_all();
        __syncthreads();
    }
    // C layout: lane 0-15 -> N=lane, M=vgpr; lane 16-31 -> N=lane-16, M=8+vgpr
    const int lidx = lane & 15, hsel = lane >> 4;
    for (int mt = 0; mt < 2; ++mt)
        for (int nt = 0; nt < 2; ++nt) {
            const int n = n0 + wn + nt * 16 + lidx;
            float badd = (bias0 ? bias0[n] : 0.0f) + (bias1 ? bias1[n] : 0.0f);
#pragma unroll
            for (int i = 0; i < 8; ++i) {
                const int m = m0 + wm + mt * 16 + hsel * 8 + i;
                Cout[(size_t)m * N + n] = acc[mt][nt][i] + badd;
            }
        }
}

// ---------------------------------------------------------------------------
// Fused LSTM step: gates = h_prev @ W_hh^T (WMMA) + gates_x[:,t,:], then
// pointwise cell update. Block tile: 64 batch rows x 128 h-cols x 4 gates.
// Grid: (B/64=4, H/128=4). h is double-buffered across steps by the host.
// Single-buffered LDS (36KB) with async-to-LDS staging (ASYNCcnt).
// ---------------------------------------------------------------------------
__global__ __launch_bounds__(256) void lstm_step_kernel(
    const unsigned short* __restrict__ hprev_, const unsigned short* __restrict__ Whh_,
    const float* __restrict__ gx, float* __restrict__ cstate,
    float* __restrict__ hs, unsigned short* __restrict__ hnext_, int t)
{
    const bf16_t* A = (const bf16_t*)hprev_;   // [B, H] bf16
    const bf16_t* W = (const bf16_t*)Whh_;     // [4H, H] bf16
    bf16_t* hnext = (bf16_t*)hnext_;
    __shared__ bf16_t As[64 * 32];
    __shared__ bf16_t Bs[4][128 * 32];         // per-gate W tiles, transposed
    const int tid  = threadIdx.x;
    const int wave = tid >> 5, lane = tid & 31;
    const int m0 = blockIdx.x * 64;            // batch rows
    const int h0 = blockIdx.y * 128;           // h columns
    const int wm = (wave >> 2) * 32;
    const int wn = (wave & 3) * 32;
    const int ar = tid >> 2, ac = (tid & 3) * 8;   // A: 8 elems (1 x b128)
    const int br = tid >> 1, bc = (tid & 1) * 16;  // W: 16 elems (2 x b128)

    const bf16_t* asrc = A + (size_t)(m0 + ar) * H_SZ + ac;
    bf16_t* al = &As[ar * 32 + ac];

    v8f acc[4][2][2] = {};

    for (int k0 = 0; k0 < H_SZ; k0 += 32) {
        __syncthreads();   // previous tile fully consumed before overwrite
        async_copy_b128<0>(asrc + k0, al);
#pragma unroll
        for (int q = 0; q < 4; ++q) {
            const bf16_t* wsrc = W + (size_t)(q * H_SZ + h0 + br) * H_SZ + k0 + bc;
            bf16_t* bl = &Bs[q][br * 32 + bc];
            async_copy_b128<0>(wsrc, bl);      // +16B chunk via IOFFSET
            async_copy_b128<16>(wsrc, bl);
        }
        wait_async_all();
        __syncthreads();
        v16bf a0 = frag_a(As, 32, wm,      0, lane);
        v16bf a1 = frag_a(As, 32, wm + 16, 0, lane);
#pragma unroll
        for (int q = 0; q < 4; ++q) {
            v16bf b0 = frag_b(Bs[q], 32, wn,      0, lane);
            v16bf b1 = frag_b(Bs[q], 32, wn + 16, 0, lane);
            acc[q][0][0] = wmma_bf16(a0, b0, acc[q][0][0]);
            acc[q][0][1] = wmma_bf16(a0, b1, acc[q][0][1]);
            acc[q][1][0] = wmma_bf16(a1, b0, acc[q][1][0]);
            acc[q][1][1] = wmma_bf16(a1, b1, acc[q][1][1]);
        }
    }
    const int lidx = lane & 15, hsel = lane >> 4;
    for (int mt = 0; mt < 2; ++mt)
        for (int nt = 0; nt < 2; ++nt) {
            const int hcol = h0 + wn + nt * 16 + lidx;
#pragma unroll
            for (int i = 0; i < 8; ++i) {
                const int m = m0 + wm + mt * 16 + hsel * 8 + i;
                const float* gxp = gx + ((size_t)m * T_SZ + t) * G4_SZ;
                float gi = acc[0][mt][nt][i] + gxp[hcol];             // i
                float gf = acc[1][mt][nt][i] + gxp[H_SZ + hcol];      // f
                float gg = acc[2][mt][nt][i] + gxp[2 * H_SZ + hcol];  // g
                float go = acc[3][mt][nt][i] + gxp[3 * H_SZ + hcol];  // o
                float ii = sigmoidf_(gi), ff = sigmoidf_(gf);
                float g  = tanhf(gg),     oo = sigmoidf_(go);
                const size_t cidx = (size_t)m * H_SZ + hcol;
                float c = ff * cstate[cidx] + ii * g;
                float h = oo * tanhf(c);
                cstate[cidx] = c;
                hs[((size_t)m * T_SZ + t) * H_SZ + hcol] = h;
                hnext[cidx] = (bf16_t)h;
            }
        }
}

// ---------------------------------------------------------------------------
// BatchNorm (training stats over batch) -> bf16 normalized activations.
// ---------------------------------------------------------------------------
__global__ __launch_bounds__(256) void bn_kernel(
    const float* __restrict__ x, const float* __restrict__ gamma,
    const float* __restrict__ beta, unsigned short* __restrict__ xn_)
{
    bf16_t* xn = (bf16_t*)xn_;
    const int j = blockIdx.x * 256 + threadIdx.x;   // column < 8192
    float s = 0.0f, s2 = 0.0f;
    for (int b = 0; b < B_SZ; ++b) {
        float v = x[(size_t)b * IN_SZ + j];
        s += v; s2 += v * v;
    }
    const float mean = s * (1.0f / B_SZ);
    const float var  = s2 * (1.0f / B_SZ) - mean * mean;
    const float inv  = rsqrtf(var + 1e-5f);
    const float ga = gamma[j] * inv;
    const float be = beta[j] - mean * ga;
    for (int b = 0; b < B_SZ; ++b) {
        float v = x[(size_t)b * IN_SZ + j];
        xn[(size_t)b * IN_SZ + j] = (bf16_t)(v * ga + be);
    }
}

__global__ void f32_to_bf16_kernel(const float* __restrict__ src,
                                   unsigned short* __restrict__ dst, size_t n)
{
    bf16_t* d = (bf16_t*)dst;
    size_t i = (size_t)blockIdx.x * blockDim.x + threadIdx.x;
    const size_t stride = (size_t)gridDim.x * blockDim.x;
    for (; i < n; i += stride) d[i] = (bf16_t)src[i];
}

__global__ void zero_init_kernel(float* __restrict__ c,
                                 unsigned short* __restrict__ h, int n)
{
    const int i = blockIdx.x * 256 + threadIdx.x;
    if (i < n) { c[i] = 0.0f; h[i] = 0; }
}

// ---------------------------------------------------------------------------
// Spatial attention: per (b,t) row of x3: sigmoid(x @ Wsa^T + b), softmax over
// feature dim, gate x, emit bf16 for the gate-projection GEMM.
// ---------------------------------------------------------------------------
__global__ __launch_bounds__(128) void spatial_attn_kernel(
    const float* __restrict__ x3, const float* __restrict__ Wsa,
    const float* __restrict__ bsa, unsigned short* __restrict__ gated_)
{
    bf16_t* gated = (bf16_t*)gated_;
    __shared__ float xs[128];
    __shared__ float red[128];
    const int row = blockIdx.x;       // b*T + t
    const int e = threadIdx.x;
    xs[e] = x3[(size_t)row * D_SZ + e];
    __syncthreads();
    float acc = bsa[e];
    const float* w = Wsa + (size_t)e * D_SZ;
#pragma unroll 4
    for (int d = 0; d < D_SZ; ++d) acc += xs[d] * w[d];
    const float a = sigmoidf_(acc);
    red[e] = a; __syncthreads();
    for (int s = 64; s > 0; s >>= 1) {
        if (e < s) red[e] = fmaxf(red[e], red[e + s]);
        __syncthreads();
    }
    const float mx = red[0]; __syncthreads();
    const float ex = __expf(a - mx);
    red[e] = ex; __syncthreads();
    for (int s = 64; s > 0; s >>= 1) {
        if (e < s) red[e] += red[e + s];
        __syncthreads();
    }
    const float alpha = ex / red[0];
    gated[(size_t)row * D_SZ + e] = (bf16_t)(xs[e] * alpha);
}

// ---------------------------------------------------------------------------
// Temporal attention + pooling + output projection. One block per batch elem.
// ---------------------------------------------------------------------------
__global__ __launch_bounds__(256) void temporal_out_kernel(
    const float* __restrict__ hs, const float* __restrict__ Wta,
    const float* __restrict__ bta, const float* __restrict__ Wout,
    float* __restrict__ out)
{
    const int TH = T_SZ * H_SZ;   // 32768
    __shared__ float part[256];
    __shared__ float logit[T_SZ];
    __shared__ float beta_s[T_SZ];
    __shared__ float pooled[H_SZ];
    const int b = blockIdx.x, tid = threadIdx.x;
    const float* hb = hs + (size_t)b * TH;
    const int t = tid >> 2, pp = tid & 3;       // 4 partial sums per t
    {
        const float* w    = Wta + (size_t)t * TH + pp * (TH / 4);
        const float* hseg = hb + pp * (TH / 4);
        float s = 0.0f;
        for (int k = 0; k < TH / 4; ++k) s += hseg[k] * w[k];
        part[tid] = s;
    }
    __syncthreads();
    if (pp == 0) {
        float l = part[tid] + part[tid + 1] + part[tid + 2] + part[tid + 3] + bta[t];
        logit[t] = fmaxf(l, 0.0f);              // relu
    }
    __syncthreads();
    if (tid == 0) {                             // softmax over T=64
        float mx = -1e30f;
        for (int i = 0; i < T_SZ; ++i) mx = fmaxf(mx, logit[i]);
        float s = 0.0f;
        for (int i = 0; i < T_SZ; ++i) { float e = __expf(logit[i] - mx); beta_s[i] = e; s += e; }
        float inv = 1.0f / s;
        for (int i = 0; i < T_SZ; ++i) beta_s[i] *= inv;
    }
    __syncthreads();
    for (int h = tid; h < H_SZ; h += 256) {
        float p = 0.0f;
        for (int tt = 0; tt < T_SZ; ++tt) p += hb[(size_t)tt * H_SZ + h] * beta_s[tt];
        pooled[h] = p;
    }
    __syncthreads();
    if (tid < O_SZ) {
        const float* w = Wout + (size_t)tid * H_SZ;
        float s = 0.0f;
        for (int h = 0; h < H_SZ; ++h) s += pooled[h] * w[h];
        out[(size_t)b * O_SZ + tid] = s;
    }
}

// ---------------------------------------------------------------------------
extern "C" void kernel_launch(void* const* d_in, const int* in_sizes, int n_in,
                              void* d_out, int out_size, void* d_ws, size_t ws_size,
                              hipStream_t stream)
{
    (void)in_sizes; (void)n_in; (void)out_size; (void)ws_size;
    const float* input    = (const float*)d_in[0];
    const float* bn_gamma = (const float*)d_in[1];
    const float* bn_beta  = (const float*)d_in[2];
    const float* W_in     = (const float*)d_in[3];
    const float* W_sa     = (const float*)d_in[4];
    const float* b_sa     = (const float*)d_in[5];
    const float* W_ih     = (const float*)d_in[6];
    const float* W_hh     = (const float*)d_in[7];
    const float* b_ih     = (const float*)d_in[8];
    const float* b_hh     = (const float*)d_in[9];
    const float* W_ta     = (const float*)d_in[10];
    const float* b_ta     = (const float*)d_in[11];
    const float* W_out    = (const float*)d_in[12];
    float* out = (float*)d_out;

    // workspace carve (~322 MB), 256B aligned
    char* p = (char*)d_ws;
    auto carve = [&](size_t bytes) -> void* {
        void* r = (void*)p; p += (bytes + 255) & ~(size_t)255; return r;
    };
    unsigned short* xn_bf    = (unsigned short*)carve((size_t)B_SZ * IN_SZ * 2);
    unsigned short* Win_bf   = (unsigned short*)carve((size_t)IN_SZ * IN_SZ * 2);
    float*          x3       = (float*)carve((size_t)B_SZ * IN_SZ * 4);
    unsigned short* Wih_bf   = (unsigned short*)carve((size_t)G4_SZ * D_SZ * 2);
    unsigned short* Whh_bf   = (unsigned short*)carve((size_t)G4_SZ * H_SZ * 2);
    unsigned short* gated_bf = (unsigned short*)carve((size_t)B_SZ * T_SZ * D_SZ * 2);
    float*          gx       = (float*)carve((size_t)B_SZ * T_SZ * G4_SZ * 4);
    unsigned short* hbuf0    = (unsigned short*)carve((size_t)B_SZ * H_SZ * 2);
    unsigned short* hbuf1    = (unsigned short*)carve((size_t)B_SZ * H_SZ * 2);
    float*          cstate   = (float*)carve((size_t)B_SZ * H_SZ * 4);
    float*          hs       = (float*)carve((size_t)B_SZ * T_SZ * H_SZ * 4);

    // weight conversions to bf16 (W_in bf16 = 128MB -> L2-resident for GEMM1)
    f32_to_bf16_kernel<<<1024, 256, 0, stream>>>(W_in, Win_bf, (size_t)IN_SZ * IN_SZ);
    f32_to_bf16_kernel<<<256, 256, 0, stream>>>(W_ih, Wih_bf, (size_t)G4_SZ * D_SZ);
    f32_to_bf16_kernel<<<512, 256, 0, stream>>>(W_hh, Whh_bf, (size_t)G4_SZ * H_SZ);
    // BatchNorm -> bf16 activations
    bn_kernel<<<IN_SZ / 256, 256, 0, stream>>>(input, bn_gamma, bn_beta, xn_bf);
    // zero LSTM state (h0, c0)
    zero_init_kernel<<<(B_SZ * H_SZ) / 256, 256, 0, stream>>>(cstate, hbuf0, B_SZ * H_SZ);

    // GEMM1: x3 = xn @ W_in^T   [256 x 8192 x 8192]  (dominant, bf16 WMMA)
    gemm_bf16_wmma<<<dim3(B_SZ / 64, IN_SZ / 128), 256, 0, stream>>>(
        xn_bf, Win_bf, x3, nullptr, nullptr, B_SZ, IN_SZ, IN_SZ);
    // spatial attention + gating -> bf16
    spatial_attn_kernel<<<B_SZ * T_SZ, 128, 0, stream>>>(x3, W_sa, b_sa, gated_bf);
    // gates_x = gated @ W_ih^T + b_ih + b_hh   [16384 x 2048 x 128]
    gemm_bf16_wmma<<<dim3((B_SZ * T_SZ) / 64, G4_SZ / 128), 256, 0, stream>>>(
        gated_bf, Wih_bf, gx, b_ih, b_hh, B_SZ * T_SZ, G4_SZ, D_SZ);
    // sequential LSTM: 64 fused WMMA+pointwise steps, h double-buffered
    unsigned short* hbufs[2] = { hbuf0, hbuf1 };
    for (int t = 0; t < T_SZ; ++t) {
        lstm_step_kernel<<<dim3(B_SZ / 64, H_SZ / 128), 256, 0, stream>>>(
            hbufs[t & 1], Whh_bf, gx, cstate, hs, hbufs[(t + 1) & 1], t);
    }
    // temporal attention, pooling, output projection
    temporal_out_kernel<<<B_SZ, 256, 0, stream>>>(hs, W_ta, b_ta, W_out, out);
}
